// GraphSAGENet_66932770341053
// MI455X (gfx1250) — compile-verified, matrix operand order checked by reading, not run on hardware
//
#include <hip/hip_runtime.h>
#include <hip/hip_bf16.h>

typedef __attribute__((ext_vector_type(16))) _Float16 v16h;
typedef __attribute__((ext_vector_type(8)))  _Float16 v8h;
typedef __attribute__((ext_vector_type(8)))  float    v8f;

#define N_NODES   50000
#define N_EDGES   800000
#define D_FEAT    128
#define HIDDEN    256
#define N_CLASSES 2

// ---------------------------------------------------------------- utilities
__global__ void zerof_kernel(float* __restrict__ p, int n) {
  int i = blockIdx.x * blockDim.x + threadIdx.x;
  int stride = gridDim.x * blockDim.x;
  for (; i < n; i += stride) p[i] = 0.0f;
}

// Wt[n][k] = (f16) ( k<128 ? W1l[k][n] : W1r[k-128][n] )   (N-major B for WMMA)
__global__ void pack_w1_kernel(const float* __restrict__ W1l,
                               const float* __restrict__ W1r,
                               _Float16* __restrict__ Wt) {
  int idx = blockIdx.x * blockDim.x + threadIdx.x;   // 256*256 threads
  int n = idx >> 8;
  int k = idx & 255;
  float w = (k < D_FEAT) ? W1l[k * HIDDEN + n] : W1r[(k - D_FEAT) * HIDDEN + n];
  Wt[idx] = (_Float16)w;
}

// ------------------------------------------------- layer-1 mean aggregation
// one wave32 per edge: lane i handles features i, i+32, i+64, i+96
__global__ __launch_bounds__(256)
void scatter1_kernel(const float* __restrict__ x, const int* __restrict__ ei,
                     float* __restrict__ agg, float* __restrict__ deg) {
  int warp = (blockIdx.x * blockDim.x + threadIdx.x) >> 5;
  int lane = threadIdx.x & 31;
  if (warp >= N_EDGES) return;
  int src = ei[warp];
  int dst = ei[N_EDGES + warp];
  const float* xs = x + (size_t)src * D_FEAT;
  float* ad = agg + (size_t)dst * D_FEAT;
#pragma unroll
  for (int f = 0; f < D_FEAT; f += 32)
    atomicAdd(&ad[f + lane], xs[f + lane]);
  if (lane == 0) atomicAdd(&deg[dst], 1.0f);
}

// A[node] = [ agg[node]/max(deg,1) | x[node] ]  converted to f16  (50000 x 256)
__global__ void pack_a_kernel(const float* __restrict__ x,
                              const float* __restrict__ agg,
                              const float* __restrict__ deg,
                              _Float16* __restrict__ A) {
  int node = blockIdx.x;
  int f = threadIdx.x;                  // 128 threads
  float inv = 1.0f / fmaxf(deg[node], 1.0f);
  A[(size_t)node * HIDDEN + f]          = (_Float16)(agg[(size_t)node * D_FEAT + f] * inv);
  A[(size_t)node * HIDDEN + D_FEAT + f] = (_Float16)x[(size_t)node * D_FEAT + f];
}

// ----------------------------------------------------------- layer-1 GEMM
// h = relu(A @ Wt^T + b1)   via v_wmma_f32_16x16x32_f16
// block = 8 waves, one 16-row M-tile per block, 2 N-tiles per wave, K=256
__global__ __launch_bounds__(256)
void gemm1_wmma_kernel(const _Float16* __restrict__ A,
                       const _Float16* __restrict__ Wt,
                       const float* __restrict__ b1,
                       float* __restrict__ h) {
  const int mbase = blockIdx.x * 16;
  const int wid   = threadIdx.x >> 5;
  const int lane  = threadIdx.x & 31;
  const int mr    = lane & 15;    // A row / B column / D column within tile
  const int kh    = lane >> 4;    // K-half selector per ISA 16-bit layout
  const int nt0   = wid * 2;      // this wave's first N-tile (of 16)

  const _Float16* arow  = A  + (size_t)(mbase + mr)          * HIDDEN;
  const _Float16* brow0 = Wt + (size_t)(nt0 * 16 + mr)       * HIDDEN;
  const _Float16* brow1 = Wt + (size_t)((nt0 + 1) * 16 + mr) * HIDDEN;

  v8f c0 = {};
  v8f c1 = {};
#pragma unroll
  for (int ks = 0; ks < HIDDEN / 32; ++ks) {
    const int k0 = ks * 32 + kh * 8;
    union { v16h v; v8h h2[2]; } ua, ub0, ub1;
    ua.h2[0]  = *(const v8h*)(arow  + k0);
    ua.h2[1]  = *(const v8h*)(arow  + k0 + 16);
    ub0.h2[0] = *(const v8h*)(brow0 + k0);
    ub0.h2[1] = *(const v8h*)(brow0 + k0 + 16);
    ub1.h2[0] = *(const v8h*)(brow1 + k0);
    ub1.h2[1] = *(const v8h*)(brow1 + k0 + 16);
    c0 = __builtin_amdgcn_wmma_f32_16x16x32_f16(false, ua.v, false, ub0.v,
                                                (short)0, c0, false, false);
    c1 = __builtin_amdgcn_wmma_f32_16x16x32_f16(false, ua.v, false, ub1.v,
                                                (short)0, c1, false, false);
  }

  // D layout: VGPR j -> row mbase + j + 8*kh ; lane -> column. Fuse bias+ReLU.
  const int col0 = nt0 * 16 + mr;
  const int col1 = col0 + 16;
  const float bias0 = b1[col0];
  const float bias1 = b1[col1];
#pragma unroll
  for (int j = 0; j < 8; ++j) {
    const size_t row = (size_t)(mbase + j + 8 * kh);
    h[row * HIDDEN + col0] = fmaxf(c0[j] + bias0, 0.0f);
    h[row * HIDDEN + col1] = fmaxf(c1[j] + bias1, 0.0f);
  }
}

// ------------------------------------------- layer-2 (project-then-scatter)
// pl = h @ W2l,  pr = h @ W2r   (50000 x 2 each); one wave per node
__global__ __launch_bounds__(256)
void proj2_kernel(const float* __restrict__ h, const float* __restrict__ W2l,
                  const float* __restrict__ W2r, float* __restrict__ pl,
                  float* __restrict__ pr) {
  int node = (blockIdx.x * blockDim.x + threadIdx.x) >> 5;
  int lane = threadIdx.x & 31;
  if (node >= N_NODES) return;
  const float* hr = h + (size_t)node * HIDDEN;
  float l0 = 0.f, l1 = 0.f, r0 = 0.f, r1 = 0.f;
#pragma unroll
  for (int f = lane; f < HIDDEN; f += 32) {
    float hv = hr[f];
    l0 += hv * W2l[f * 2 + 0];
    l1 += hv * W2l[f * 2 + 1];
    r0 += hv * W2r[f * 2 + 0];
    r1 += hv * W2r[f * 2 + 1];
  }
#pragma unroll
  for (int off = 16; off > 0; off >>= 1) {
    l0 += __shfl_down(l0, off, 32);
    l1 += __shfl_down(l1, off, 32);
    r0 += __shfl_down(r0, off, 32);
    r1 += __shfl_down(r1, off, 32);
  }
  if (lane == 0) {
    pl[node * 2 + 0] = l0; pl[node * 2 + 1] = l1;
    pr[node * 2 + 0] = r0; pr[node * 2 + 1] = r1;
  }
}

// aggp[dst] += pl[src]  — only 8 bytes/edge thanks to linearity of segment_sum
__global__ void scatter_p_kernel(const int* __restrict__ ei,
                                 const float* __restrict__ pl,
                                 float* __restrict__ aggp) {
  int e = blockIdx.x * blockDim.x + threadIdx.x;
  if (e >= N_EDGES) return;
  int src = ei[e];
  int dst = ei[N_EDGES + e];
  atomicAdd(&aggp[dst * 2 + 0], pl[src * 2 + 0]);
  atomicAdd(&aggp[dst * 2 + 1], pl[src * 2 + 1]);
}

// out = sigmoid(aggp/max(deg,1) + pr + b2)
__global__ void final2_kernel(const float* __restrict__ aggp,
                              const float* __restrict__ pr,
                              const float* __restrict__ deg,
                              const float* __restrict__ b2,
                              float* __restrict__ out) {
  int idx = blockIdx.x * blockDim.x + threadIdx.x;
  if (idx >= N_NODES * N_CLASSES) return;
  int node = idx >> 1;
  int c = idx & 1;
  float inv = 1.0f / fmaxf(deg[node], 1.0f);
  float v = aggp[idx] * inv + pr[idx] + b2[c];
  out[idx] = 1.0f / (1.0f + __expf(-v));
}

// --------------------------------------------------------------- launcher
extern "C" void kernel_launch(void* const* d_in, const int* in_sizes, int n_in,
                              void* d_out, int out_size, void* d_ws, size_t ws_size,
                              hipStream_t stream) {
  const float* x   = (const float*)d_in[0];
  const int*   ei  = (const int*)d_in[1];
  const float* W1l = (const float*)d_in[2];
  const float* W1r = (const float*)d_in[3];
  const float* b1  = (const float*)d_in[4];
  const float* W2l = (const float*)d_in[5];
  const float* W2r = (const float*)d_in[6];
  const float* b2  = (const float*)d_in[7];

  float* out = (float*)d_out;                       // [50000 x 2]
  float* h   = (float*)d_out + N_NODES * N_CLASSES; // embeddings [50000 x 256]

  // workspace layout (bytes)
  char* ws = (char*)d_ws;
  float*    agg1 = (float*)(ws + 0);                         // 25,600,000 B
  _Float16* Af16 = (_Float16*)(ws + 25600000);               // 25,600,000 B
  float*    deg  = (float*)(ws + 51200000);                  //    200,000 B
  _Float16* Wt   = (_Float16*)(ws + 51400000);               //    131,072 B
  float*    pl   = (float*)(ws + 51531072);                  //    400,000 B
  float*    aggp = (float*)(ws + 51931072);                  //    400,000 B
  float*    pr   = (float*)(ws + 52331072);                  //    400,000 B

  // zero accumulators (fresh every call: graph-replay safe)
  zerof_kernel<<<2048, 256, 0, stream>>>(agg1, N_NODES * D_FEAT);
  zerof_kernel<<<128, 256, 0, stream>>>(deg, N_NODES);
  zerof_kernel<<<128, 256, 0, stream>>>(aggp, N_NODES * N_CLASSES);

  // pack transposed, concatenated layer-1 weights to f16
  pack_w1_kernel<<<(HIDDEN * HIDDEN) / 256, 256, 0, stream>>>(W1l, W1r, Wt);

  // layer-1 neighbor aggregation (one wave per edge)
  scatter1_kernel<<<N_EDGES / 8, 256, 0, stream>>>(x, ei, agg1, deg);

  // build A = [mean | x] in f16
  pack_a_kernel<<<N_NODES, D_FEAT, 0, stream>>>(x, agg1, deg, Af16);

  // h = relu(A @ W + b1) via WMMA; written directly into d_out embeddings slot
  gemm1_wmma_kernel<<<N_NODES / 16, 256, 0, stream>>>(Af16, Wt, b1, h);

  // layer-2: project to 2 classes first, then scatter the 2-vectors
  proj2_kernel<<<(N_NODES * 32) / 256, 256, 0, stream>>>(h, W2l, W2r, pl, pr);
  scatter_p_kernel<<<(N_EDGES + 255) / 256, 256, 0, stream>>>(ei, pl, aggp);
  final2_kernel<<<(N_NODES * N_CLASSES + 255) / 256, 256, 0, stream>>>(aggp, pr, deg, b2, out);
}
